// GPT_29592324669825
// MI455X (gfx1250) — matrix operand order repaired and checked by reading, not compiled, not run on hardware
//
#include <hip/hip_runtime.h>
#include <hip/hip_bf16.h>
#include <math.h>

// ---------- model dims ----------
constexpr int Cd  = 768;
constexpr int Td  = 1024;
constexpr int Bd  = 2;
constexpr int Hd  = 12;
constexpr int HDd = 64;
constexpr int Ld  = 12;
constexpr int Vd  = 50257;
constexpr int BT  = Bd * Td;          // 2048 activation rows

// ---------- WMMA tile config ----------
constexpr int GBM = 128;              // block tile M (4 wave-rows of 32)
constexpr int GBN = 64;               // block tile N (2 wave-cols of 32)
constexpr int GBK = 32;               // K step (16x16x32 bf16 WMMA)
constexpr int LDP = 36;               // LDS row stride (ushorts): 72B -> 8B aligned pairs
constexpr int ABUF = GBM * LDP;       // ushorts per A double-buffer slice
constexpr int BBUF = GBN * LDP;       // ushorts per B double-buffer slice

typedef __attribute__((ext_vector_type(16))) __bf16 v16bf;
typedef __attribute__((ext_vector_type(2)))  __bf16 v2bf;
typedef __attribute__((ext_vector_type(8)))  float  v8f;

__device__ __forceinline__ unsigned int pack_bf16(float lo, float hi) {
  v2bf p;
  p.x = (__bf16)lo;                   // v_cvt_pk_bf16_f32 on gfx1250
  p.y = (__bf16)hi;
  return __builtin_bit_cast(unsigned int, p);
}

union Frag { unsigned int u[8]; v16bf v; };

// =====================================================================
// Generic batched GEMM: C[z] = epilogue(A[z] @ B[z])
// MODE 0: (+bias)   1: (+bias+residual)   2: (+bias, exact GELU)
// MODE 3: causal mask + scale (attention scores)
// NCHK=true  -> N divisible by GBN, no bounds checks compiled in.
// Batch z decomposed as (z/nH, z%nH) with separate strides (head views).
// Double-buffered LDS; staging split into load-phase then convert/store
// phase so global_load_b128s clause together and overlap the converts.
// =====================================================================
template<int MODE, bool TRANSB, bool NCHK>
__global__ __launch_bounds__(256)
void gemm_wmma(const float* __restrict__ A,  long aB, long aH, int lda,
               const float* __restrict__ Bm, long bB, long bH, int ldb,
               const float* __restrict__ bias,
               const float* __restrict__ resid,
               float* __restrict__ Cm, long cB, long cH, int ldc,
               int M, int N, int K, int nH, float scale)
{
  __shared__ unsigned short As[2][GBM][LDP];   // bf16 bits, [m][k]
  __shared__ unsigned short Bs[2][GBN][LDP];   // bf16 bits, [n][k] (k pairs contiguous)

  const int t    = threadIdx.x;
  const int lane = t & 31;
  const int wid  = t >> 5;
  const int tm   = (wid & 3) * 32;    // wave row block within GBM
  const int tn   = (wid >> 2) * 32;   // wave col block within GBN

  const int z  = blockIdx.z;
  const int zb = z / nH, zh = z % nH;
  A  += (long)zb * aB + (long)zh * aH;
  Bm += (long)zb * bB + (long)zh * bH;
  Cm += (long)zb * cB + (long)zh * cH;

  const int m0 = blockIdx.y * GBM;
  const int n0 = blockIdx.x * GBN;

  // ---- precompute per-thread staging pointers (advance by GBK per step) ----
  const float* aPtr[4];
  unsigned short* aLds[4];
  #pragma unroll
  for (int i = 0; i < 4; ++i) {
    int idx = i * 256 + t;                 // 0..1023
    int r   = idx >> 3;                    // row 0..127
    int c4  = (idx & 7) * 4;               // k group of 4
    aPtr[i] = A + (long)(m0 + r) * lda + c4;
    aLds[i] = &As[0][r][c4];
  }
  const float* bPtr[4];
  unsigned short* bLds[4];
  bool bOk[2] = {true, true};
  if (TRANSB) {
    #pragma unroll
    for (int i = 0; i < 2; ++i) {
      int idx = i * 256 + t;               // 0..511
      int n   = idx >> 3;                  // 0..63
      int c4  = (idx & 7) * 4;
      bOk[i]  = NCHK || (n0 + n) < N;
      bPtr[i] = Bm + (long)(n0 + n) * ldb + c4;
      bLds[i] = &Bs[0][n][c4];
    }
  } else {
    #pragma unroll
    for (int i = 0; i < 2; ++i) {
      int p  = i * 256 + t;                // 0..511
      int n2 = (p & 31) * 2;               // n pair
      int k2 = (p >> 5) * 2;               // k pair
      bPtr[2 * i]     = Bm + (long)k2 * ldb + (n0 + n2);
      bPtr[2 * i + 1] = Bm + (long)(k2 + 1) * ldb + (n0 + n2);
      bLds[2 * i]     = &Bs[0][n2][k2];
      bLds[2 * i + 1] = &Bs[0][n2 + 1][k2];
    }
  }

  // ---- tile stager: phase 1 = all global loads, phase 2 = convert+LDS ----
  auto stage = [&](int buf) {
    float4 av[4];
    #pragma unroll
    for (int i = 0; i < 4; ++i) {          // A loads (clause together)
      av[i] = *(const float4*)aPtr[i];
      aPtr[i] += GBK;
    }
    if (TRANSB) {
      float4 bv[2];
      #pragma unroll
      for (int i = 0; i < 2; ++i) {        // B loads
        bv[i] = make_float4(0.f, 0.f, 0.f, 0.f);
        if (bOk[i]) bv[i] = *(const float4*)bPtr[i];
        bPtr[i] += GBK;
      }
      #pragma unroll
      for (int i = 0; i < 4; ++i)          // A convert + store
        *(uint2*)(aLds[i] + buf * ABUF) =
            make_uint2(pack_bf16(av[i].x, av[i].y), pack_bf16(av[i].z, av[i].w));
      #pragma unroll
      for (int i = 0; i < 2; ++i)          // B convert + store
        *(uint2*)(bLds[i] + buf * BBUF) =
            make_uint2(pack_bf16(bv[i].x, bv[i].y), pack_bf16(bv[i].z, bv[i].w));
    } else {
      float2 b0[2], b1[2];
      #pragma unroll
      for (int i = 0; i < 2; ++i) {        // B loads (2x2 patches)
        b0[i] = *(const float2*)bPtr[2 * i];
        b1[i] = *(const float2*)bPtr[2 * i + 1];
        bPtr[2 * i]     += (long)GBK * ldb;
        bPtr[2 * i + 1] += (long)GBK * ldb;
      }
      #pragma unroll
      for (int i = 0; i < 4; ++i)          // A convert + store
        *(uint2*)(aLds[i] + buf * ABUF) =
            make_uint2(pack_bf16(av[i].x, av[i].y), pack_bf16(av[i].z, av[i].w));
      #pragma unroll
      for (int i = 0; i < 2; ++i) {        // B convert + store
        *(unsigned int*)(bLds[2 * i]     + buf * BBUF) = pack_bf16(b0[i].x, b1[i].x);
        *(unsigned int*)(bLds[2 * i + 1] + buf * BBUF) = pack_bf16(b0[i].y, b1[i].y);
      }
    }
  };

  v8f acc[2][2] = {};

  stage(0);
  __syncthreads();

  const int nsteps = K / GBK;
  for (int s = 0; s < nsteps; ++s) {
    const int cur = s & 1;
    // stage next tile into the other buffer while computing on `cur`
    if (s + 1 < nsteps) {
      stage(cur ^ 1);
      __builtin_prefetch(aPtr[0], 0, 0);   // hint tile after next -> global_prefetch_b8
    }

    // ---- gather fragments in contiguous k order (k = 16*(lane/16) + element), same for A and B ----
    Frag fa[2], fb[2];
    {
      const int rbase = tm + (lane & 15);
      const int kb    = (lane < 16) ? 0 : 16;  // A: lanes0-15 K=0-15; lanes16-31 K=16-31
      #pragma unroll
      for (int ss = 0; ss < 2; ++ss) {
        #pragma unroll
        for (int j = 0; j < 8; ++j) {
          int k = kb + 2 * j;
          fa[ss].u[j] = *(const unsigned int*)&As[cur][rbase + ss * 16][k];
        }
      }
      const int cbase = tn + (lane & 15);
      const int kb2   = (lane < 16) ? 0 : 16;  // B: lanes0-15 K=0-15; lanes16-31 K=16-31
      #pragma unroll
      for (int ss = 0; ss < 2; ++ss) {
        #pragma unroll
        for (int j = 0; j < 8; ++j) {
          fb[ss].u[j] = *(const unsigned int*)&Bs[cur][cbase + ss * 16][kb2 + 2 * j];
        }
      }
    }

    #pragma unroll
    for (int sm = 0; sm < 2; ++sm)
      #pragma unroll
      for (int sn = 0; sn < 2; ++sn)
        acc[sm][sn] = __builtin_amdgcn_wmma_f32_16x16x32_bf16(
            false, fa[sm].v, false, fb[sn].v, (short)0, acc[sm][sn], false, false);

    __syncthreads();   // consumers of `cur` done AND producers of `cur^1` done
  }

  // ---- epilogue: C/D layout: VGPR r -> M = r + 8*(lane>=16), N = lane&15 ----
  const int row_half = (lane >> 4) * 8;
  #pragma unroll
  for (int sn = 0; sn < 2; ++sn) {
    const int col = n0 + tn + sn * 16 + (lane & 15);
    if (NCHK || col < N) {
      const float bv = (MODE != 3 && bias) ? bias[col] : 0.0f;
      float* cp = &Cm[(long)(m0 + tm + row_half) * ldc + col];
      const float* rp = (MODE == 1) ? &resid[(long)(m0 + tm + row_half) * ldc + col] : nullptr;
      #pragma unroll
      for (int sm = 0; sm < 2; ++sm) {
        #pragma unroll
        for (int r = 0; r < 8; ++r) {
          const long off = (long)(sm * 16 + r) * ldc;
          float v = acc[sm][sn][r] + bv;
          if (MODE == 1) v += rp[off];
          if (MODE == 2) v = 0.5f * v * (1.0f + erff(v * 0.70710678118654752440f));
          if (MODE == 3) {
            const int gm = m0 + tm + sm * 16 + row_half + r;
            v = (col <= gm) ? acc[sm][sn][r] * scale : -1e9f;
          }
          cp[off] = v;
        }
      }
    }
  }
}

// =====================================================================
// Token + position embedding: x[bt] = wte[idx[bt]] + wpe[bt % T]
// =====================================================================
__global__ __launch_bounds__(256)
void embed_k(const int* __restrict__ idx, const float* __restrict__ wte,
             const float* __restrict__ wpe, float* __restrict__ x)
{
  const int bt = blockIdx.x;
  const long tok = idx[bt];
  const float* we = wte + tok * Cd;
  const float* wp = wpe + (long)(bt % Td) * Cd;
  float* xr = x + (long)bt * Cd;
  for (int c = threadIdx.x; c < Cd; c += 256) xr[c] = we[c] + wp[c];
}

// =====================================================================
// LayerNorm (one block per row)
// =====================================================================
__global__ __launch_bounds__(256)
void layernorm_k(const float* __restrict__ x, const float* __restrict__ g,
                 const float* __restrict__ b, float* __restrict__ y)
{
  __shared__ float red[256];
  const float* xr = x + (long)blockIdx.x * Cd;
  float s = 0.f, sq = 0.f;
  for (int c = threadIdx.x; c < Cd; c += 256) { float v = xr[c]; s += v; sq += v * v; }
  red[threadIdx.x] = s; __syncthreads();
  for (int o = 128; o > 0; o >>= 1) { if (threadIdx.x < o) red[threadIdx.x] += red[threadIdx.x + o]; __syncthreads(); }
  const float mean = red[0] / Cd; __syncthreads();
  red[threadIdx.x] = sq; __syncthreads();
  for (int o = 128; o > 0; o >>= 1) { if (threadIdx.x < o) red[threadIdx.x] += red[threadIdx.x + o]; __syncthreads(); }
  const float var = red[0] / Cd - mean * mean;
  const float inv = rsqrtf(var + 1e-5f);
  float* yr = y + (long)blockIdx.x * Cd;
  for (int c = threadIdx.x; c < Cd; c += 256) yr[c] = (xr[c] - mean) * inv * g[c] + b[c];
}

// =====================================================================
// Row softmax over T columns (one block per row)
// =====================================================================
__global__ __launch_bounds__(256)
void softmax_k(float* __restrict__ s)
{
  __shared__ float red[256];
  float* r = s + (long)blockIdx.x * Td;
  float mx = -3.4e38f;
  for (int c = threadIdx.x; c < Td; c += 256) mx = fmaxf(mx, r[c]);
  red[threadIdx.x] = mx; __syncthreads();
  for (int o = 128; o > 0; o >>= 1) { if (threadIdx.x < o) red[threadIdx.x] = fmaxf(red[threadIdx.x], red[threadIdx.x + o]); __syncthreads(); }
  mx = red[0]; __syncthreads();
  float sum = 0.f;
  for (int c = threadIdx.x; c < Td; c += 256) { float e = __expf(r[c] - mx); r[c] = e; sum += e; }
  red[threadIdx.x] = sum; __syncthreads();
  for (int o = 128; o > 0; o >>= 1) { if (threadIdx.x < o) red[threadIdx.x] += red[threadIdx.x + o]; __syncthreads(); }
  const float inv = 1.0f / red[0];
  for (int c = threadIdx.x; c < Td; c += 256) r[c] *= inv;
}

// =====================================================================
// Host orchestration
// =====================================================================
extern "C" void kernel_launch(void* const* d_in, const int* in_sizes, int n_in,
                              void* d_out, int out_size, void* d_ws, size_t ws_size,
                              hipStream_t stream)
{
  (void)in_sizes; (void)n_in; (void)out_size; (void)ws_size;
  const int*   idx    = (const int*)  d_in[0];
  const float* wte    = (const float*)d_in[1];
  const float* wpe    = (const float*)d_in[2];
  const float* ln1_g  = (const float*)d_in[3];
  const float* ln1_b  = (const float*)d_in[4];
  const float* attn_w = (const float*)d_in[5];
  const float* attn_b = (const float*)d_in[6];
  const float* proj_w = (const float*)d_in[7];
  const float* proj_b = (const float*)d_in[8];
  const float* ln2_g  = (const float*)d_in[9];
  const float* ln2_b  = (const float*)d_in[10];
  const float* fc_w   = (const float*)d_in[11];
  const float* fc_b   = (const float*)d_in[12];
  const float* fp_w   = (const float*)d_in[13];
  const float* fp_b   = (const float*)d_in[14];
  const float* lnf_g  = (const float*)d_in[15];
  const float* lnf_b  = (const float*)d_in[16];
  float* out = (float*)d_out;

  // workspace layout (fp32)
  float* x      = (float*)d_ws;                                   // [BT, C]
  float* h      = x      + (size_t)BT * Cd;                       // [BT, C]
  float* qkv    = h      + (size_t)BT * Cd;                       // [BT, 3C]
  float* scores = qkv    + (size_t)BT * 3 * Cd;                   // [B*H, T, T]
  float* o      = scores + (size_t)Bd * Hd * Td * Td;             // [BT, C]
  float* act4   = o      + (size_t)BT * Cd;                       // [BT, 4C]

  const dim3 blk(256);
  embed_k<<<BT, blk, 0, stream>>>(idx, wte, wpe, x);

  for (int l = 0; l < Ld; ++l) {
    const float* aw  = attn_w + (size_t)l * Cd * 3 * Cd;
    const float* ab  = attn_b + (size_t)l * 3 * Cd;
    const float* pw  = proj_w + (size_t)l * Cd * Cd;
    const float* pb  = proj_b + (size_t)l * Cd;
    const float* fw  = fc_w   + (size_t)l * Cd * 4 * Cd;
    const float* fb  = fc_b   + (size_t)l * 4 * Cd;
    const float* fpw = fp_w   + (size_t)l * 4 * Cd * Cd;
    const float* fpb = fp_b   + (size_t)l * Cd;

    // ---- LN1 ----
    layernorm_k<<<BT, blk, 0, stream>>>(x, ln1_g + (size_t)l * Cd, ln1_b + (size_t)l * Cd, h);

    // ---- qkv = h @ attn_w + attn_b : [2048,768]x[768,2304] ----
    gemm_wmma<0, false, true><<<dim3(3 * Cd / GBN, BT / GBM, 1), blk, 0, stream>>>(
        h, 0, 0, Cd,   aw, 0, 0, 3 * Cd,   ab, nullptr,
        qkv, 0, 0, 3 * Cd,   BT, 3 * Cd, Cd, 1, 1.0f);

    // ---- scores = causal(Q @ K^T * 1/8) : 24 batched [1024,64]x[64,1024] ----
    gemm_wmma<3, true, true><<<dim3(Td / GBN, Td / GBM, Bd * Hd), blk, 0, stream>>>(
        qkv,        (long)Td * 3 * Cd, 3 * HDd, 3 * Cd,     // Q view
        qkv + HDd,  (long)Td * 3 * Cd, 3 * HDd, 3 * Cd,     // K view (transposed access)
        nullptr, nullptr,
        scores, (long)Hd * Td * Td, (long)Td * Td, Td,
        Td, Td, HDd, Hd, 0.125f);

    softmax_k<<<Bd * Hd * Td, blk, 0, stream>>>(scores);

    // ---- o = att @ V : 24 batched [1024,1024]x[1024,64] ----
    gemm_wmma<0, false, true><<<dim3(HDd / GBN, Td / GBM, Bd * Hd), blk, 0, stream>>>(
        scores,        (long)Hd * Td * Td, (long)Td * Td, Td,
        qkv + 2 * HDd, (long)Td * 3 * Cd,  3 * HDd,       3 * Cd,   // V view
        nullptr, nullptr,
        o, (long)Td * Cd, HDd, Cd,
        Td, HDd, Td, Hd, 1.0f);

    // ---- x = x + o @ proj_w + proj_b ----
    gemm_wmma<1, false, true><<<dim3(Cd / GBN, BT / GBM, 1), blk, 0, stream>>>(
        o, 0, 0, Cd,   pw, 0, 0, Cd,   pb, x,
        x, 0, 0, Cd,   BT, Cd, Cd, 1, 1.0f);

    // ---- LN2 ----
    layernorm_k<<<BT, blk, 0, stream>>>(x, ln2_g + (size_t)l * Cd, ln2_b + (size_t)l * Cd, h);

    // ---- act4 = gelu(h @ fc_w + fc_b) : [2048,768]x[768,3072] ----
    gemm_wmma<2, false, true><<<dim3(4 * Cd / GBN, BT / GBM, 1), blk, 0, stream>>>(
        h, 0, 0, Cd,   fw, 0, 0, 4 * Cd,   fb, nullptr,
        act4, 0, 0, 4 * Cd,   BT, 4 * Cd, Cd, 1, 1.0f);

    // ---- x = x + act4 @ fp_w + fp_b : [2048,3072]x[3072,768] ----
    gemm_wmma<1, false, true><<<dim3(Cd / GBN, BT / GBM, 1), blk, 0, stream>>>(
        act4, 0, 0, 4 * Cd,   fpw, 0, 0, Cd,   fpb, x,
        x, 0, 0, Cd,   BT, Cd, 4 * Cd, 1, 1.0f);
  }

  // ---- final LN + tied-embedding logits: [2048,768]x[768,50257] (B=wte^T) ----
  layernorm_k<<<BT, blk, 0, stream>>>(x, lnf_g, lnf_b, h);
  gemm_wmma<0, true, false><<<dim3((Vd + GBN - 1) / GBN, BT / GBM, 1), blk, 0, stream>>>(
      h, 0, 0, Cd,   wte, 0, 0, Cd,   nullptr, nullptr,
      out, 0, 0, Vd,   BT, Vd, Cd, 1, 1.0f);
}
